// TtCausalSelfAttention_10746008174885
// MI455X (gfx1250) — matrix operand order repaired and checked by reading, not compiled, not run on hardware
//
#include <hip/hip_runtime.h>

typedef __attribute__((ext_vector_type(16))) _Float16 v16h;
typedef __attribute__((ext_vector_type(8)))  _Float16 v8h;
typedef __attribute__((ext_vector_type(8)))  float    v8f;

#define B_  4
#define T_  1024
#define C_  1024
#define H_  16
#define HD_ 64
#define C3_ 3072

struct Frag {
    union { v16h v; v8h p[2]; _Float16 h[16]; };
};

// ---------------------------------------------------------------------------
// f32 -> f16 cast (grid-stride)
// ---------------------------------------------------------------------------
__global__ void cvt_f32_f16(const float* __restrict__ in,
                            _Float16* __restrict__ out, int n) {
    int i = blockIdx.x * blockDim.x + threadIdx.x;
    int stride = gridDim.x * blockDim.x;
    for (; i < n; i += stride) out[i] = (_Float16)in[i];
}

// ---------------------------------------------------------------------------
// C[M,N] = A[M,K] * W[N,K]^T + bias  (both operands K-contiguous)
// block = 256 thr (8 waves). Block tile: 128 rows x 64 cols.
// Wave tile: 16 rows x 64 cols. Software pipeline with two statically-named
// fragment sets (ping/pong) so everything stays in VGPRs; K must be a
// multiple of 64 (it is: 1024).
// ---------------------------------------------------------------------------
__global__ __launch_bounds__(256) void gemm_bias_wmma(
    const _Float16* __restrict__ A,    // [M,K]
    const _Float16* __restrict__ W,    // [N,K]
    const float*    __restrict__ bias, // [N]
    _Float16* __restrict__ outH,       // f16 output (or null)
    float*    __restrict__ outF,       // f32 output (or null)
    int M, int N, int K)
{
    const int lane = threadIdx.x & 31;
    const int wv   = threadIdx.x >> 5;
    const int g    = lane >> 4;   // half-wave group
    const int n    = lane & 15;   // lane within group
    const int rowBase = blockIdx.y * 128 + wv * 16;
    const int colBase = blockIdx.x * 64;

    v8f acc[4];
#pragma unroll
    for (int t = 0; t < 4; ++t)
#pragma unroll
        for (int e = 0; e < 8; ++e) acc[t][e] = 0.0f;

    // A fragment 16x32: lane holds row (rowBase+n); K = g*16..+15
    const _Float16* aPtr = A + (size_t)(rowBase + n) * K + g * 16;
    // B fragment 32x16: lane n holds col (colBase+nt*16+n) = W row; K = g*16..+15
    const _Float16* wPtr[4];
#pragma unroll
    for (int nt = 0; nt < 4; ++nt)
        wPtr[nt] = W + (size_t)(colBase + nt * 16 + n) * K + g * 16;

    Frag aA, aB, bA[4], bB[4];

#define LOAD_SET(aF, bF, kk)                                      \
    do {                                                          \
        aF.p[0] = *(const v8h*)(aPtr + (kk));                     \
        aF.p[1] = *(const v8h*)(aPtr + (kk) + 8);                 \
        _Pragma("unroll")                                         \
        for (int nt = 0; nt < 4; ++nt) {                          \
            bF[nt].p[0] = *(const v8h*)(wPtr[nt] + (kk));         \
            bF[nt].p[1] = *(const v8h*)(wPtr[nt] + (kk) + 8);     \
        }                                                         \
    } while (0)

#define MMA_SET(aF, bF)                                           \
    do {                                                          \
        _Pragma("unroll")                                         \
        for (int nt = 0; nt < 4; ++nt)                            \
            acc[nt] = __builtin_amdgcn_wmma_f32_16x16x32_f16(     \
                false, aF.v, false, bF[nt].v, (short)0, acc[nt],  \
                false, false);                                    \
    } while (0)

    LOAD_SET(aA, bA, 0);
    for (int kk = 0; kk < K; kk += 64) {
        LOAD_SET(aB, bB, kk + 32);           // in flight over first MMA set
        __builtin_prefetch(aPtr + kk + 64, 0, 1);
        MMA_SET(aA, bA);
        if (kk + 64 < K)
            LOAD_SET(aA, bA, kk + 64);       // in flight over second MMA set
        MMA_SET(aB, bB);
    }
#undef LOAD_SET
#undef MMA_SET

#pragma unroll
    for (int nt = 0; nt < 4; ++nt) {
        const int col = colBase + nt * 16 + n;
        const float bv = bias[col];
#pragma unroll
        for (int r = 0; r < 8; ++r) {
            const int row = rowBase + r + g * 8;   // C/D layout: reg r -> row r (+8 hi half)
            const float val = acc[nt][r] + bv;
            if (outH) outH[(size_t)row * N + col] = (_Float16)val;
            else      outF[(size_t)row * N + col] = val;
        }
    }
}

// ---------------------------------------------------------------------------
// Causal flash attention over qkv[B,T,3C] (f16) -> y[B,T,C] (f16)
// grid = (B*H, T/128); 8 waves/block; wave owns a 16-row q tile.
// V chunks are staged to LDS with async global->LDS B128 copies (ASYNCcnt).
// ---------------------------------------------------------------------------
__global__ __launch_bounds__(256) void attn_wmma(
    const _Float16* __restrict__ qkv,
    _Float16* __restrict__ y)
{
    __shared__ __align__(16) _Float16 pbuf[8][16][40]; // per-wave P transpose staging
    __shared__ __align__(16) _Float16 vbuf[8][32][72]; // per-wave V chunk (row stride 144B)

    const int lane = threadIdx.x & 31;
    const int wv   = threadIdx.x >> 5;
    const int g    = lane >> 4;
    const int n    = lane & 15;
    const int b = blockIdx.x >> 4;
    const int h = blockIdx.x & 15;
    const int qBase = blockIdx.y * 128 + wv * 16;

    // Q fragments for d-chunks {0,32}, pre-scaled by 1/sqrt(hd) = 0.125
    Frag qf[2];
    {
        const _Float16* qp =
            qkv + (size_t)(b * T_ + qBase + n) * C3_ + h * HD_;
#pragma unroll
        for (int c = 0; c < 2; ++c) {
            qf[c].p[0] = *(const v8h*)(qp + c * 32 + g * 16);
            qf[c].p[1] = *(const v8h*)(qp + c * 32 + g * 16 + 8);
#pragma unroll
            for (int e = 0; e < 16; ++e)
                qf[c].h[e] = (_Float16)((float)qf[c].h[e] * 0.125f);
        }
    }

    // per-lane async-copy source/dest for V staging: lane owns k-row (kc+lane)
    const unsigned ldsRow = (unsigned)(uintptr_t)(&vbuf[wv][lane][0]);
    unsigned long long gRow = (unsigned long long)(uintptr_t)(
        qkv + (size_t)(b * T_ + lane) * C3_ + 2 * C_ + h * HD_);
    const unsigned long long gRowStep = (unsigned long long)32 * C3_ * 2;

    v8f o[4];
    float m[8], l[8];
#pragma unroll
    for (int t = 0; t < 4; ++t)
#pragma unroll
        for (int e = 0; e < 8; ++e) o[t][e] = 0.0f;
#pragma unroll
    for (int r = 0; r < 8; ++r) { m[r] = -3.0e38f; l[r] = 0.0f; }

    const int qEnd = qBase + 15;
    for (int kc = 0; kc <= qEnd; kc += 32, gRow += gRowStep) {
        // ---- kick off async V stage: 32 rows x 64 halves, 16B per lane per op ----
#pragma unroll
        for (int i = 0; i < 8; ++i) {
            asm volatile("global_load_async_to_lds_b128 %0, %1, off"
                         :: "v"(ldsRow + i * 16), "v"(gRow + i * 16)
                         : "memory");
        }

        // ---- S = Q * K^T for 32 k-columns (two 16x16 C tiles) ----
        v8f s[2];
#pragma unroll
        for (int i = 0; i < 2; ++i)
#pragma unroll
            for (int e = 0; e < 8; ++e) s[i][e] = 0.0f;
#pragma unroll
        for (int i = 0; i < 2; ++i) {
#pragma unroll
            for (int c = 0; c < 2; ++c) {
                Frag kf;  // B frag: lane n = K row (kc+i*16+n); d = c*32 + g*16 .. +15
                const _Float16* kp =
                    qkv + (size_t)(b * T_ + kc + i * 16 + n) * C3_ +
                    C_ + h * HD_ + c * 32 + g * 16;
                kf.p[0] = *(const v8h*)(kp);
                kf.p[1] = *(const v8h*)(kp + 8);
                s[i] = __builtin_amdgcn_wmma_f32_16x16x32_f16(
                    false, qf[c].v, false, kf.v, (short)0, s[i], false, false);
            }
        }

        // ---- causal mask + online softmax (row groups match C-layout halves) ----
#pragma unroll
        for (int r = 0; r < 8; ++r) {
            const int row = qBase + r + g * 8;
#pragma unroll
            for (int i = 0; i < 2; ++i) {
                const int col = kc + i * 16 + n;
                if (col > row) s[i][r] = -3.0e38f;
            }
            float mx = fmaxf(s[0][r], s[1][r]);
#pragma unroll
            for (int off = 8; off >= 1; off >>= 1)
                mx = fmaxf(mx, __shfl_xor(mx, off, 32));
            const float mn = fmaxf(m[r], mx);
            const float alpha = __expf(m[r] - mn);
            const float p0 = __expf(s[0][r] - mn);
            const float p1 = __expf(s[1][r] - mn);
            pbuf[wv][r + g * 8][n]      = (_Float16)p0;
            pbuf[wv][r + g * 8][16 + n] = (_Float16)p1;
            float sum = p0 + p1;
#pragma unroll
            for (int off = 8; off >= 1; off >>= 1)
                sum += __shfl_xor(sum, off, 32);
            l[r] = l[r] * alpha + sum;
            m[r] = mn;
#pragma unroll
            for (int nt = 0; nt < 4; ++nt) o[nt][r] *= alpha;
        }

        // ---- reload P as A-fragment (transpose via LDS) ----
        Frag pf;
        const _Float16* pp = &pbuf[wv][n][0];
        pf.p[0] = *(const v8h*)(pp + g * 16);
        pf.p[1] = *(const v8h*)(pp + g * 16 + 8);

        // ---- wait for async V stage, then O += P * V ----
        asm volatile("s_wait_asynccnt 0x0" ::: "memory");
#pragma unroll
        for (int nt = 0; nt < 4; ++nt) {
            Frag vf;  // B frag: lane n = column d = nt*16+n; k row = g*16+e
#pragma unroll
            for (int e = 0; e < 16; ++e)
                vf.h[e] = vbuf[wv][g * 16 + e][nt * 16 + n];
            o[nt] = __builtin_amdgcn_wmma_f32_16x16x32_f16(
                false, pf.v, false, vf.v, (short)0, o[nt], false, false);
        }
    }

    // ---- normalize + store y[b, row, h*64 + d] ----
#pragma unroll
    for (int r = 0; r < 8; ++r) {
        const float inv = 1.0f / l[r];
        const int row = qBase + r + g * 8;
#pragma unroll
        for (int nt = 0; nt < 4; ++nt)
            y[(size_t)(b * T_ + row) * C_ + h * HD_ + nt * 16 + n] =
                (_Float16)(o[nt][r] * inv);
    }
}

// ---------------------------------------------------------------------------
extern "C" void kernel_launch(void* const* d_in, const int* in_sizes, int n_in,
                              void* d_out, int out_size, void* d_ws, size_t ws_size,
                              hipStream_t stream) {
    (void)in_sizes; (void)n_in; (void)out_size; (void)ws_size;
    const float* x     = (const float*)d_in[0];   // [4,1024,1024]
    const float* wAttn = (const float*)d_in[1];   // [3072,1024]
    const float* bAttn = (const float*)d_in[2];   // [3072]
    const float* wProj = (const float*)d_in[3];   // [1024,1024]
    const float* bProj = (const float*)d_in[4];   // [1024]
    float* out = (float*)d_out;                   // [4,1024,1024]

    char* ws = (char*)d_ws;
    _Float16* xh   = (_Float16*)(ws);                               //  8 MiB
    _Float16* wah  = (_Float16*)(ws + (size_t) 8 * 1024 * 1024);    //  6 MiB
    _Float16* wph  = (_Float16*)(ws + (size_t)14 * 1024 * 1024);    //  2 MiB
    _Float16* qkvh = (_Float16*)(ws + (size_t)16 * 1024 * 1024);    // 24 MiB
    _Float16* yh   = (_Float16*)(ws + (size_t)40 * 1024 * 1024);    //  8 MiB

    cvt_f32_f16<<<2048, 256, 0, stream>>>(x,     xh,  B_ * T_ * C_);
    cvt_f32_f16<<<2048, 256, 0, stream>>>(wAttn, wah, C3_ * C_);
    cvt_f32_f16<<<1024, 256, 0, stream>>>(wProj, wph, C_ * C_);

    // qkv = x @ w_attn^T + b_attn   (M=4096, N=3072, K=1024) -> f16
    gemm_bias_wmma<<<dim3(C3_ / 64, (B_ * T_) / 128), 256, 0, stream>>>(
        xh, wah, bAttn, qkvh, nullptr, B_ * T_, C3_, C_);

    // causal flash attention -> yh (f16)
    attn_wmma<<<dim3(B_ * H_, T_ / 128), 256, 0, stream>>>(qkvh, yh);

    // out = y @ w_proj^T + b_proj   (M=4096, N=1024, K=1024) -> f32
    gemm_bias_wmma<<<dim3(C_ / 64, (B_ * T_) / 128), 256, 0, stream>>>(
        yh, wph, bProj, nullptr, out, B_ * T_, C_, C_);
}
